// MLPDecoder_38878043963402
// MI455X (gfx1250) — matrix operand order repaired and checked by reading, not compiled, not run on hardware
//
#include <hip/hip_runtime.h>

// ---------------------------------------------------------------------------
// Edge-MLP link-prediction decoder for MI455X (gfx1250, wave32, WMMA).
//   out[e] = relu( (h[u]⊙h[v]) @ W1 + b1 ) @ W2 + b2
// Main GEMM runs on v_wmma_f32_16x16x32_bf16 (f32 accumulate).
// ---------------------------------------------------------------------------

typedef __attribute__((ext_vector_type(16))) __bf16 bf16x16;
typedef __attribute__((ext_vector_type(8)))  float  f32x8;

struct alignas(16) Quad { unsigned int x, y, z, w; };
struct Frag32 { Quad lo, hi; };   // 32 bytes == one 16-element bf16 fragment

__device__ __forceinline__ unsigned short f2bf(float f) {
    // round-to-nearest-even f32 -> bf16
    unsigned int u = __float_as_uint(f);
    u += 0x7fffu + ((u >> 16) & 1u);
    return (unsigned short)(u >> 16);
}

// ---------------------------------------------------------------------------
// Prep: pack W1 (256x256 f32, row=k, col=n) into bf16 B-fragments.
// Fragment (t,c) covers N-tile t (16 cols) and K-chunk c (32 deep).
// Lane layout (dense 32x16 bf16 B): lanes 0-15 hold K=c*32+0..15 of column
// n=t*16+lane; lanes 16-31 hold K=c*32+16..31 of column n=t*16+(lane-16).
// Element i within a lane is K-consecutive, so a lane's 16 bf16 = 32B blob.
// dst index = ((t*8+c)*32 + lane)*16 + i   (65536 bf16 = 128 KB total)
// ---------------------------------------------------------------------------
__global__ __launch_bounds__(256) void pack_w1_bf16(
    const float* __restrict__ W1, unsigned short* __restrict__ dst)
{
    int idx  = blockIdx.x * 256 + threadIdx.x;   // 0..65535
    int i    = idx & 15;
    int lane = (idx >> 4) & 31;
    int frag = idx >> 9;                         // 0..127
    int c    = frag & 7;
    int t    = frag >> 3;
    int k    = c * 32 + ((lane >> 4) << 4) + i;
    int n    = t * 16 + (lane & 15);
    dst[idx] = f2bf(W1[k * 256 + n]);
}

// ---------------------------------------------------------------------------
// Main kernel: 64 edges / block, 256 threads = 8 waves.
// ---------------------------------------------------------------------------
#define DFEAT 256
#define MB    64
#define LPAD  264   // LDS row stride in bf16 elems (256 + 8 pad -> no bank conflicts)

__global__ __launch_bounds__(256) void edge_mlp_wmma(
    const float*          __restrict__ h,
    const int*            __restrict__ edges,   // [E,2] flattened
    const float*          __restrict__ b1,
    const float*          __restrict__ W2,      // [256,1]
    const float*          __restrict__ b2,
    const unsigned short* __restrict__ w1p,     // packed bf16 W1 fragments
    float*                __restrict__ out,
    int E)
{
    __shared__ alignas(16) unsigned short had[MB * LPAD];  // Hadamard tile, bf16
    __shared__ int   eIdx[2 * MB];
    __shared__ float oacc[MB];

    const int tid  = threadIdx.x;
    const int base = blockIdx.x * MB;

    // ---- phase 0: stage edge indices (clamped) + zero output accumulator ----
    if (tid < 2 * MB) {
        long gi   = 2L * base + tid;
        long gmax = 2L * E - 1;
        eIdx[tid] = edges[gi > gmax ? gmax : gi];
    }
    if (tid < MB) oacc[tid] = 0.0f;
    __syncthreads();

    // ---- phase 1: gather + Hadamard -> bf16 in LDS ----
    {
        const int rgrp = tid >> 6;          // 4 rows per pass
        const int kk   = (tid & 63) << 2;   // 4 floats per thread (b128 loads)
        for (int j = 0; j < 16; ++j) {
            const int  row = (j << 2) + rgrp;
            const long u   = (long)eIdx[2 * row];
            const long v   = (long)eIdx[2 * row + 1];
            const float4 a  = *(const float4*)(h + u * DFEAT + kk);
            const float4 bb = *(const float4*)(h + v * DFEAT + kk);
            unsigned int p0 = (unsigned int)f2bf(a.x * bb.x) |
                              ((unsigned int)f2bf(a.y * bb.y) << 16);
            unsigned int p1 = (unsigned int)f2bf(a.z * bb.z) |
                              ((unsigned int)f2bf(a.w * bb.w) << 16);
            unsigned int* dst = (unsigned int*)&had[row * LPAD + kk];
            dst[0] = p0; dst[1] = p1;
        }
    }
    __syncthreads();

    // ---- phase 2: [64 x 256] @ [256 x 32-per-wave] via WMMA bf16 ----
    const int lane  = tid & 31;
    const int wv    = tid >> 5;        // wave id 0..7
    const int lcol  = lane & 15;
    const int khalf = lane >> 4;       // 0: K-low half, 1: K-high half
    const int n0    = wv * 32 + lcol;  // this lane's column in N-tile 0
    const int n1    = n0 + 16;         // ... in N-tile 1

    f32x8 acc[4][2];
    {
        const float b1a = b1[n0], b1b = b1[n1];   // bias folded into C init
#pragma unroll
        for (int m = 0; m < 4; ++m)
#pragma unroll
            for (int r = 0; r < 8; ++r) { acc[m][0][r] = b1a; acc[m][1][r] = b1b; }
    }

    const int   t0  = wv * 2;
    const Quad* w1q = (const Quad*)w1p;   // 16-byte units; fragment = 64 Quads

    for (int c = 0; c < 8; ++c) {
        // B fragments: one contiguous 32B blob per lane (2x global_load_b128)
        Frag32 B0, B1;
        const Quad* p0 = w1q + (((t0    ) * 8 + c) * 64 + lane * 2);
        const Quad* p1 = w1q + (((t0 + 1) * 8 + c) * 64 + lane * 2);
        B0.lo = p0[0]; B0.hi = p0[1];
        B1.lo = p1[0]; B1.hi = p1[1];
        const bf16x16 vB0 = __builtin_bit_cast(bf16x16, B0);
        const bf16x16 vB1 = __builtin_bit_cast(bf16x16, B1);
#pragma unroll
        for (int m = 0; m < 4; ++m) {
            // A fragment from LDS: lanes 0-15 take K {0..15},
            // lanes 16-31 take K {16..31} of row m*16+lcol.
            Frag32 A;
            const unsigned short* ap =
                &had[(m * 16 + lcol) * LPAD + c * 32 + khalf * 16];
            A.lo = *(const Quad*)(ap);
            A.hi = *(const Quad*)(ap + 8);
            const bf16x16 vA = __builtin_bit_cast(bf16x16, A);
            acc[m][0] = __builtin_amdgcn_wmma_f32_16x16x32_bf16(
                false, vA, false, vB0, (short)0, acc[m][0], false, false);
            acc[m][1] = __builtin_amdgcn_wmma_f32_16x16x32_bf16(
                false, vA, false, vB1, (short)0, acc[m][1], false, false);
        }
    }

    // ---- phase 3: fused ReLU -> @W2 -> row reduction ----
    // C/D layout: VGPR r holds row (r + 8*khalf) for column lcol of each tile.
    {
        const float w2a = W2[n0], w2b = W2[n1];
#pragma unroll
        for (int m = 0; m < 4; ++m) {
            float s[8];
#pragma unroll
            for (int r = 0; r < 8; ++r)
                s[r] = fmaxf(acc[m][0][r], 0.0f) * w2a +
                       fmaxf(acc[m][1][r], 0.0f) * w2b;
            // butterfly-reduce over the 16 column-lanes (stays within halves)
#pragma unroll
            for (int off = 1; off < 16; off <<= 1)
#pragma unroll
                for (int r = 0; r < 8; ++r)
                    s[r] += __shfl_xor(s[r], off, 32);
            if (lcol == 0) {
                const int rowbase = m * 16 + khalf * 8;
#pragma unroll
                for (int r = 0; r < 8; ++r)
                    atomicAdd(&oacc[rowbase + r], s[r]);   // ds_add_f32
            }
        }
    }
    __syncthreads();

    if (tid < MB) {
        const int e = base + tid;
        if (e < E) out[e] = oacc[tid] + b2[0];
    }
}

// ---------------------------------------------------------------------------
// Host-side launcher (graph-capture safe: only kernel launches on `stream`).
// Inputs: h[100000,256] f32, edges[E,2] int, W1[256,256] f32, b1[256] f32,
//         W2[256,1] f32, b2[1] f32.  Output: [E] f32.
// ---------------------------------------------------------------------------
extern "C" void kernel_launch(void* const* d_in, const int* in_sizes, int n_in,
                              void* d_out, int out_size, void* d_ws, size_t ws_size,
                              hipStream_t stream)
{
    const float* h     = (const float*)d_in[0];
    const int*   edges = (const int*)  d_in[1];
    const float* W1    = (const float*)d_in[2];
    const float* b1    = (const float*)d_in[3];
    const float* W2    = (const float*)d_in[4];
    const float* b2    = (const float*)d_in[5];
    float*       out   = (float*)d_out;
    const int    E     = in_sizes[1] / 2;

    unsigned short* w1p = (unsigned short*)d_ws;   // 65536 bf16 = 128 KB

    // one-shot repack of W1 into WMMA B-fragment layout (bf16)
    pack_w1_bf16<<<256, 256, 0, stream>>>(W1, w1p);

    const int nblk = (E + MB - 1) / MB;
    edge_mlp_wmma<<<nblk, 256, 0, stream>>>(h, edges, b1, W2, b2, w1p, out, E);
}
